// GRUSlowExtractor_40003325394985
// MI455X (gfx1250) — matrix operand order repaired and checked by reading, not compiled, not run on hardware
//
#include <hip/hip_runtime.h>

// ---------------------------------------------------------------------------
// Fused 2-layer GRU (H=64) + MLP head for B=256, T=2048, D=1 on gfx1250.
// One workgroup (12 waves, 384 threads) owns 16 batch rows for the whole
// time loop. Recurrent/input GEMMs use V_WMMA_F32_16X16X32_F16 with weight
// B-fragments resident in VGPRs; state lives in LDS (f32 master + f16 mirror).
// Nonlinearities use v_exp_f32 + v_rcp_f32 (TRANS ops, co-execute with XDL).
// ---------------------------------------------------------------------------

typedef _Float16 v16h __attribute__((ext_vector_type(16)));
typedef _Float16 v8h  __attribute__((ext_vector_type(8)));
typedef float    v8f  __attribute__((ext_vector_type(8)));

#define GRU_B   256
#define GRU_T   2048
#define GRU_H   64
#define GRU_G   192   // 3*H

// f16 weight pack offsets (in halves) inside d_ws
#define OFF_WHH0 0
#define OFF_WIH1 12288
#define OFF_WHH1 24576
#define OFF_WH1  36864

// sigmoid via v_exp_f32 + v_rcp_f32 (no IEEE-div expansion)
__device__ __forceinline__ float sigm(float v) {
  return __builtin_amdgcn_rcpf(1.0f + __expf(-v));
}
// tanh(x) = 2*sigmoid(2x) - 1  (exp underflow -> rcp(inf)=0 -> -1: correct)
__device__ __forceinline__ float tanh_fast(float v) {
  return fmaf(2.0f, sigm(2.0f * v), -1.0f);
}

// A fragment (16x32 f16, MxK) from an LDS tile stored row-major [16][64].
// Lane l<16 : M=l,   e0..15 = K(c*32+0..15)
// Lane l>=16: M=l-16,e0..15 = K(c*32+16..31)
__device__ __forceinline__ v16h loadA(const _Float16* hm, int lane, int c) {
  const int row = lane & 15;
  const int sel = lane >> 4;
  const _Float16* p = hm + row * 64 + c * 32 + sel * 16;
  union { v16h v; v8h h[2]; } u;
  u.h[0] = *(const v8h*)(p);
  u.h[1] = *(const v8h*)(p + 8);
  return u.v;
}

// C = A(hm[16][64]) x B (two resident K-chunks), f32 accumulate, K=64.
__device__ __forceinline__ v8f mm64(const _Float16* hm, int lane,
                                    v16h B0, v16h B1) {
  v8f c = {};
  v16h a0 = loadA(hm, lane, 0);
  v16h a1 = loadA(hm, lane, 1);
  c = __builtin_amdgcn_wmma_f32_16x16x32_f16(false, a0, false, B0,
                                             (short)0, c, false, false);
  c = __builtin_amdgcn_wmma_f32_16x16x32_f16(false, a1, false, B1,
                                             (short)0, c, false, false);
  return c;
}

// Scatter a 16x16 f32 C fragment into LDS [16][stride].
// VGPR j, lanes 0-15 -> (M=j, N=lane); lanes 16-31 -> (M=j+8, N=lane-16).
__device__ __forceinline__ void storeC(float* g, int stride, v8f c,
                                       int wave, int lane) {
  const int n   = lane & 15;
  const int sel = lane >> 4;
#pragma unroll
  for (int j = 0; j < 8; ++j)
    g[(j + 8 * sel) * stride + wave * 16 + n] = c[j];
}

// ---- prep: pack W_hh0 / W_ih1 / W_hh1 / Wh1 to f16 in workspace ----------
__global__ void gru_pack_weights(const float* __restrict__ whh0,
                                 const float* __restrict__ wih1,
                                 const float* __restrict__ whh1,
                                 const float* __restrict__ wh1,
                                 _Float16* __restrict__ out) {
  int i = blockIdx.x * 256 + threadIdx.x;   // grid covers 12288
  if (i < 12288) {
    out[OFF_WHH0 + i] = (_Float16)whh0[i];
    out[OFF_WIH1 + i] = (_Float16)wih1[i];
    out[OFF_WHH1 + i] = (_Float16)whh1[i];
  }
  if (i < 4096) out[OFF_WH1 + i] = (_Float16)wh1[i];
}

// ---- fused GRU+head kernel: grid = 16 blocks x 384 threads ----------------
__global__ __launch_bounds__(384, 1)
void gru_fused_kernel(const float* __restrict__ x,
                      const float* __restrict__ h0in,
                      const float* __restrict__ wih0,
                      const float* __restrict__ bih0,
                      const float* __restrict__ bhh0,
                      const float* __restrict__ bih1,
                      const float* __restrict__ bhh1,
                      const float* __restrict__ bh1,
                      const float* __restrict__ wh2,
                      const float* __restrict__ bh2,
                      const _Float16* __restrict__ wf16,
                      float* __restrict__ y_out,
                      float* __restrict__ hT_out) {
  const int tid  = threadIdx.x;
  const int wave = tid >> 5;
  const int lane = tid & 31;
  const int b0   = blockIdx.x * 16;       // global batch base of this WG
  const bool ew  = (tid < 256);           // waves 0-7: elementwise work

  __shared__ float    sh0[16 * 64];       // layer0 hidden, f32 master
  __shared__ float    sh1[16 * 64];       // layer1 hidden, f32 master
  __shared__ _Float16 sh0h[16 * 64];      // f16 mirrors (WMMA A input)
  __shared__ _Float16 sh1h[16 * 64];
  __shared__ float    gA[16 * GRU_G];     // gate scratch (gh0 / gx1)
  __shared__ float    gB[16 * GRU_G];     // gate scratch (gh1)
  __shared__ float    sht[16 * 64];       // head hidden (post-ReLU)
  __shared__ float    sxs[16 * 128];      // staged x chunk
  __shared__ float    sys[16 * 128];      // staged y chunk
  __shared__ float    p_wih0[GRU_G], p_bih0[GRU_G], p_bhh0[GRU_G];
  __shared__ float    p_bih1[GRU_G], p_bhh1[GRU_G];
  __shared__ float    p_bh1[64], p_wh2[64];
  __shared__ float    p_bh2;

  if (tid < GRU_G) {
    p_wih0[tid] = wih0[tid]; p_bih0[tid] = bih0[tid]; p_bhh0[tid] = bhh0[tid];
    p_bih1[tid] = bih1[tid]; p_bhh1[tid] = bhh1[tid];
  }
  if (tid < 64) { p_bh1[tid] = bh1[tid]; p_wh2[tid] = wh2[tid]; }
  if (tid == 0) p_bh2 = bh2[0];
  if (ew) {
#pragma unroll
    for (int r = 0; r < 4; ++r) {
      int i = tid + 256 * r;
      int b = i >> 6, j = i & 63;
      float v0 = h0in[(b0 + b) * 64 + j];
      float v1 = h0in[GRU_B * 64 + (b0 + b) * 64 + j];
      sh0[i] = v0; sh0h[i] = (_Float16)v0;
      sh1[i] = v1; sh1h[i] = (_Float16)v1;
    }
  }

  // Resident WMMA B-fragments: wave w owns output columns [16w, 16w+16).
  // B[k][n] = W[ow+n][k]; lane covers K(khalf*16..+16) of each 32-K chunk.
  const int n     = lane & 15;
  const int khalf = lane >> 4;
  const int o     = wave * 16 + n;
  const _Float16* whh0h = wf16 + OFF_WHH0;
  const _Float16* wih1h = wf16 + OFF_WIH1;
  const _Float16* whh1h = wf16 + OFF_WHH1;
  const _Float16* wh1h  = wf16 + OFF_WH1;
  v16h Bhh0_0 = *(const v16h*)(whh0h + o * 64 +      khalf * 16);
  v16h Bhh0_1 = *(const v16h*)(whh0h + o * 64 + 32 + khalf * 16);
  v16h Bih1_0 = *(const v16h*)(wih1h + o * 64 +      khalf * 16);
  v16h Bih1_1 = *(const v16h*)(wih1h + o * 64 + 32 + khalf * 16);
  v16h Bhh1_0 = *(const v16h*)(whh1h + o * 64 +      khalf * 16);
  v16h Bhh1_1 = *(const v16h*)(whh1h + o * 64 + 32 + khalf * 16);
  v16h Bh1_0 = {}, Bh1_1 = {};
  if (wave < 4) {                         // head has only 64 outputs
    Bh1_0 = *(const v16h*)(wh1h + o * 64 +      khalf * 16);
    Bh1_1 = *(const v16h*)(wh1h + o * 64 + 32 + khalf * 16);
  }
  __syncthreads();

  for (int t = 0; t < GRU_T; ++t) {
    const int tm = t & 127;
    if (tm == 0) {                        // stage 128 timesteps of x
      __syncthreads();
      if (ew) {
#pragma unroll
        for (int r = 0; r < 8; ++r) {
          int i = tid + 256 * r;
          int b = i >> 7, j = i & 127;
          sxs[i] = x[(b0 + b) * GRU_T + t + j];
        }
      }
      __syncthreads();
    }

    // ---- layer 0: gh0 = h0 @ W_hh0^T  (12 waves x 16x16 tiles) ----
    { v8f c = mm64(sh0h, lane, Bhh0_0, Bhh0_1); storeC(gA, GRU_G, c, wave, lane); }
    __syncthreads();

    // ---- layer 0 gates (gx0 computed on the fly: D==1) ----
    if (ew) {
#pragma unroll
      for (int r = 0; r < 4; ++r) {
        int i = tid + 256 * r;
        int b = i >> 6, j = i & 63;
        float xv = sxs[b * 128 + tm];
        float xr = fmaf(xv, p_wih0[j],       p_bih0[j]);
        float xz = fmaf(xv, p_wih0[64 + j],  p_bih0[64 + j]);
        float xn = fmaf(xv, p_wih0[128 + j], p_bih0[128 + j]);
        float hr = gA[b * GRU_G + j]       + p_bhh0[j];
        float hz = gA[b * GRU_G + 64 + j]  + p_bhh0[64 + j];
        float hn = gA[b * GRU_G + 128 + j] + p_bhh0[128 + j];
        float rg = sigm(xr + hr);
        float zg = sigm(xz + hz);
        float ng = tanh_fast(fmaf(rg, hn, xn));
        float hv = fmaf(zg, sh0[i] - ng, ng);   // (1-z)*n + z*h
        sh0[i] = hv; sh0h[i] = (_Float16)hv;
      }
    }
    __syncthreads();

    // ---- layer 1: gx1 = z0_t @ W_ih1^T, gh1 = h1 @ W_hh1^T ----
    { v8f cx = mm64(sh0h, lane, Bih1_0, Bih1_1); storeC(gA, GRU_G, cx, wave, lane);
      v8f ch = mm64(sh1h, lane, Bhh1_0, Bhh1_1); storeC(gB, GRU_G, ch, wave, lane); }
    __syncthreads();

    // ---- layer 1 gates ----
    if (ew) {
#pragma unroll
      for (int r = 0; r < 4; ++r) {
        int i = tid + 256 * r;
        int b = i >> 6, j = i & 63;
        float xr = gA[b * GRU_G + j]       + p_bih1[j];
        float xz = gA[b * GRU_G + 64 + j]  + p_bih1[64 + j];
        float xn = gA[b * GRU_G + 128 + j] + p_bih1[128 + j];
        float hr = gB[b * GRU_G + j]       + p_bhh1[j];
        float hz = gB[b * GRU_G + 64 + j]  + p_bhh1[64 + j];
        float hn = gB[b * GRU_G + 128 + j] + p_bhh1[128 + j];
        float rg = sigm(xr + hr);
        float zg = sigm(xz + hz);
        float ng = tanh_fast(fmaf(rg, hn, xn));
        float hv = fmaf(zg, sh1[i] - ng, ng);
        sh1[i] = hv; sh1h[i] = (_Float16)hv;
      }
    }
    __syncthreads();

    // ---- head GEMM: relu(z1 @ Wh1^T + bh1) on waves 0..3 ----
    if (wave < 4) {
      v8f c = mm64(sh1h, lane, Bh1_0, Bh1_1);
      const int sel = lane >> 4;
      float bb = p_bh1[wave * 16 + n];
#pragma unroll
      for (int j = 0; j < 8; ++j) {
        float v = c[j] + bb;
        sht[(j + 8 * sel) * 64 + wave * 16 + n] = v > 0.0f ? v : 0.0f;
      }
    }
    __syncthreads();

    // ---- head output: y = ht @ Wh2^T + bh2 (16 dots of length 64) ----
    if (tid < 16) {
      float acc = p_bh2;
#pragma unroll 8
      for (int j = 0; j < 64; ++j)
        acc = fmaf(sht[tid * 64 + j], p_wh2[j], acc);
      sys[tid * 128 + tm] = acc;
    }

    if (tm == 127) {                      // flush 128 staged y values
      __syncthreads();
      if (ew) {
#pragma unroll
        for (int r = 0; r < 8; ++r) {
          int i = tid + 256 * r;
          int b = i >> 7, j = i & 127;
          y_out[(b0 + b) * GRU_T + (t - 127) + j] = sys[i];
        }
      }
    }
  }

  __syncthreads();
  if (ew) {                               // final hidden states hT[2,B,H]
#pragma unroll
    for (int r = 0; r < 4; ++r) {
      int i = tid + 256 * r;
      int b = i >> 6, j = i & 63;
      hT_out[(b0 + b) * 64 + j]              = sh0[i];
      hT_out[GRU_B * 64 + (b0 + b) * 64 + j] = sh1[i];
    }
  }
}

extern "C" void kernel_launch(void* const* d_in, const int* in_sizes, int n_in,
                              void* d_out, int out_size, void* d_ws, size_t ws_size,
                              hipStream_t stream) {
  (void)in_sizes; (void)n_in; (void)out_size; (void)ws_size;
  const float* x    = (const float*)d_in[0];
  const float* h0   = (const float*)d_in[1];
  const float* wih0 = (const float*)d_in[2];
  const float* whh0 = (const float*)d_in[3];
  const float* bih0 = (const float*)d_in[4];
  const float* bhh0 = (const float*)d_in[5];
  const float* wih1 = (const float*)d_in[6];
  const float* whh1 = (const float*)d_in[7];
  const float* bih1 = (const float*)d_in[8];
  const float* bhh1 = (const float*)d_in[9];
  const float* wh1  = (const float*)d_in[10];
  const float* bh1  = (const float*)d_in[11];
  const float* wh2  = (const float*)d_in[12];
  const float* bh2  = (const float*)d_in[13];

  _Float16* wf16 = (_Float16*)d_ws;       // 40960 halves = 80 KB
  float* y  = (float*)d_out;              // [B,T,1]
  float* hT = y + GRU_B * GRU_T;          // [2,B,H]

  gru_pack_weights<<<48, 256, 0, stream>>>(whh0, wih1, whh1, wh1, wf16);
  gru_fused_kernel<<<GRU_B / 16, 384, 0, stream>>>(
      x, h0, wih0, bih0, bhh0, bih1, bhh1, bh1, wh2, bh2,
      (const _Float16*)wf16, y, hT);
}
